// ConvApproximation_16501264351609
// MI455X (gfx1250) — matrix operand order repaired and checked by reading, not compiled, not run on hardware
//
#include <hip/hip_runtime.h>
#include <math.h>

// Problem constants (from reference)
#define B_      16
#define N_      20000
#define C_      256
#define D_      64
#define M_      64
#define NPAD    20480          // N padded to multiple of KCHUNK (zero-filled)
#define KCHUNK  512
#define NCHUNKS 40             // NPAD / KCHUNK
#define NRMZ    0.35355339059327373f   // 1/d^{1/4} * 1/sqrt(tau)
#define RATIO   0.125f                 // 1/sqrt(M)
#define EPSF    1e-4f

typedef __attribute__((ext_vector_type(16))) __bf16 v16bf;
typedef __attribute__((ext_vector_type(8)))  __bf16 v8bf;
typedef __attribute__((ext_vector_type(8)))  float  v8f;

static __device__ __forceinline__ v16bf frag_cat(v8bf lo, v8bf hi) {
  v16bf r;
#pragma unroll
  for (int i = 0; i < 8; ++i) { r[i] = lo[i]; r[8 + i] = hi[i]; }
  return r;
}

// A-matrix fragment (16x32 bf16), source row-major [16 rows of M, K contiguous].
// ISA layout: lanes 0-15 row M=lane, K 0-7 (v0..3) and 16-23 (v4..7);
//             lanes 16-31 row M=lane-16, K 8-15 and 24-31.
static __device__ __forceinline__ v16bf load_a_frag(const __bf16* base, int lda, int lane) {
  int row  = lane & 15;
  int half = (lane >> 4) & 1;
  const __bf16* p = base + (size_t)row * lda + half * 8;
  v8bf lo = *(const v8bf*)(p);
  v8bf hi = *(const v8bf*)(p + 16);
  return frag_cat(lo, hi);
}

// B-matrix fragment (32x16 bf16), source column-major: element (k,c) at src[c*ldb + k].
// ISA layout: lanes 0-15 col N=lane, K 0-7 (v0..3) and 16-23 (v4..7);
//             lanes 16-31 col N=lane-16, K 8-15 and 24-31.
static __device__ __forceinline__ v16bf load_b_frag(const __bf16* base, int ldb, int lane) {
  int col  = lane & 15;
  int koff = ((lane >> 4) & 1) * 8;
  const __bf16* p = base + (size_t)col * ldb + koff;
  v8bf lo = *(const v8bf*)(p);
  v8bf hi = *(const v8bf*)(p + 16);
  return frag_cat(lo, hi);
}

static __device__ __forceinline__ v8f wmma_bf16(v16bf a, v16bf b, v8f c) {
  return __builtin_amdgcn_wmma_f32_16x16x32_bf16(false, a, false, b, (short)0, c, false, false);
}

// ---------------- Feature-map kernels ----------------

// k_dash[n][m] = (nv2[n]*NRMZ) . proj[m];  diag_k[n]; per-block max.
__global__ __launch_bounds__(256) void fk1_kdash(const float* __restrict__ nv2,
                                                 const float* __restrict__ proj,
                                                 float* __restrict__ kdash,
                                                 float* __restrict__ diagk,
                                                 float* __restrict__ bmax) {
  __shared__ float sh[256];
  int t = threadIdx.x;
  int row = blockIdx.x * 4 + (t >> 6);
  int m   = t & 63;
  const float* dn = nv2 + (size_t)row * D_;
  const float* pr = proj + (size_t)m * D_;
  float acc = 0.f, sq = 0.f;
#pragma unroll 8
  for (int d = 0; d < D_; ++d) {
    float v = dn[d] * NRMZ;
    acc += v * pr[d];
    sq  += v * v;
  }
  kdash[(size_t)row * M_ + m] = acc;
  if (m == 0) diagk[row] = 0.5f * sq;
  sh[t] = acc;
  __syncthreads();
#pragma unroll
  for (int s = 128; s > 0; s >>= 1) {
    if (t < s) sh[t] = fmaxf(sh[t], sh[t + s]);
    __syncthreads();
  }
  if (t == 0) bmax[blockIdx.x] = sh[0];
}

__global__ __launch_bounds__(256) void fk2_max(const float* __restrict__ bmax,
                                               float* __restrict__ kstab, int n) {
  __shared__ float sh[256];
  int t = threadIdx.x;
  float mx = -3.4e38f;
  for (int i = t; i < n; i += 256) mx = fmaxf(mx, bmax[i]);
  sh[t] = mx;
  __syncthreads();
#pragma unroll
  for (int s = 128; s > 0; s >>= 1) {
    if (t < s) sh[t] = fmaxf(sh[t], sh[t + s]);
    __syncthreads();
  }
  if (t == 0) kstab[0] = sh[0];
}

// k' = ratio*(exp(kdash - diag - kstab)+eps): f32 to d_out, bf16 transposed [M][NPAD]
// (padding rows -> 0), plus per-block partial column sums.
__global__ __launch_bounds__(256) void fk3_kprime(const float* __restrict__ kdash,
                                                  const float* __restrict__ diagk,
                                                  const float* __restrict__ kstab,
                                                  float* __restrict__ kout,
                                                  __bf16* __restrict__ kpT,
                                                  float* __restrict__ kpart) {
  __shared__ float sh[256];
  int t = threadIdx.x;
  size_t idx = (size_t)blockIdx.x * 256 + t;
  int n = (int)(idx >> 6);
  int m = (int)(idx & 63);
  float stab = kstab[0];
  float val = 0.f;
  if (n < N_) {
    float e = RATIO * (expf(kdash[idx] - diagk[n] - stab) + EPSF);
    kout[idx] = e;
    val = e;
  }
  kpT[(size_t)m * NPAD + n] = (__bf16)val;
  sh[t] = val;
  __syncthreads();
  if (t < 64) kpart[(size_t)blockIdx.x * 64 + t] = sh[t] + sh[t + 64] + sh[t + 128] + sh[t + 192];
}

__global__ __launch_bounds__(256) void fk4_ksum(const float* __restrict__ kpart,
                                                float* __restrict__ ksum) {
  __shared__ float sh[256];
  int t = threadIdx.x;
  int m = t & 63, sl = t >> 6;
  float s = 0.f;
  for (int p = sl; p < (NPAD * 64) / 256; p += 4) s += kpart[(size_t)p * 64 + m];
  sh[t] = s;
  __syncthreads();
  if (t < 64) ksum[t] = sh[t] + sh[t + 64] + sh[t + 128] + sh[t + 192];
}

// q' features (row-max stabilizer), bf16 row-major copy, and denom^{-1}.
__global__ __launch_bounds__(256) void fq_qprime(const float* __restrict__ nv1,
                                                 const float* __restrict__ proj,
                                                 const float* __restrict__ ksum,
                                                 float* __restrict__ qout,
                                                 __bf16* __restrict__ qpb,
                                                 float* __restrict__ dinv) {
  __shared__ float sh[256];
  __shared__ float sdiag[4];
  int t = threadIdx.x;
  int rl = t >> 6, m = t & 63;
  int row = blockIdx.x * 4 + rl;
  const float* dn = nv1 + (size_t)row * D_;
  const float* pr = proj + (size_t)m * D_;
  float acc = 0.f, sq = 0.f;
#pragma unroll 8
  for (int d = 0; d < D_; ++d) {
    float v = dn[d] * NRMZ;
    acc += v * pr[d];
    sq  += v * v;
  }
  if (m == 0) sdiag[rl] = 0.5f * sq;
  sh[t] = acc;
  __syncthreads();
#pragma unroll
  for (int s = 32; s > 0; s >>= 1) {
    if (m < s) sh[t] = fmaxf(sh[t], sh[t + s]);
    __syncthreads();
  }
  float rmax = sh[rl * 64];
  float q = RATIO * (expf(acc - sdiag[rl] - rmax) + EPSF);
  qout[(size_t)row * M_ + m] = q;
  qpb[(size_t)row * M_ + m] = (__bf16)q;
  __syncthreads();
  sh[t] = q * ksum[m];
  __syncthreads();
#pragma unroll
  for (int s = 32; s > 0; s >>= 1) {
    if (m < s) sh[t] += sh[t + s];
    __syncthreads();
  }
  if (m == 0) dinv[row] = 1.0f / sh[t];
}

// ---------------- kv = k'^T x  (split-K WMMA, partials) ----------------
// grid (NCHUNKS, B). Block = 256 threads = 8 waves; wave w: mtile=w&3, cgroup=w>>2.
__global__ __launch_bounds__(256) void kv_gemm(const float* __restrict__ x,
                                               const __bf16* __restrict__ kpT,
                                               float* __restrict__ kvp) {
  __shared__ __bf16 xt[C_ * 32];  // [c][k], 16 KB, bf16 transposed x tile
  int b = blockIdx.y, chunk = blockIdx.x;
  int t = threadIdx.x, lane = t & 31, w = t >> 5;
  int mtile = w & 3, cg = w >> 2;
  int n0 = chunk * KCHUNK;

  // Loader decomposition: each thread owns a 4(c) x 8(k) subtile.
  // Global reads stay coalesced (64 consecutive threads cover one 1KB x-row);
  // per column the 8 k-values are contiguous in the transposed LDS layout,
  // so stores are ds_store_b128.
  int c4 = (t & 63) * 4;   // column base 0..252
  int kb = (t >> 6) * 8;   // k base: 0,8,16,24

  v8f acc[8];
#pragma unroll
  for (int i = 0; i < 8; ++i) acc[i] = (v8f){0.f, 0.f, 0.f, 0.f, 0.f, 0.f, 0.f, 0.f};

  for (int ks = 0; ks < KCHUNK; ks += 32) {
    // Prefetch next k-tile of x into cache (global_prefetch_b8).
    if (ks + 32 < KCHUNK) {
      int prow = n0 + ks + 32 + kb;
      if (prow < N_)
        __builtin_prefetch(x + ((size_t)b * N_ + prow) * C_ + c4, 0, 1);
    }
    // Stage x[b][n0+ks .. +32)[0..256) into LDS, transposed + converted to bf16.
    float rc[4][8];
#pragma unroll
    for (int g = 0; g < 8; ++g) {
      int row = n0 + ks + kb + g;
      float4 v = make_float4(0.f, 0.f, 0.f, 0.f);
      if (row < N_) v = *(const float4*)(x + ((size_t)b * N_ + row) * C_ + c4);
      rc[0][g] = v.x; rc[1][g] = v.y; rc[2][g] = v.z; rc[3][g] = v.w;
    }
#pragma unroll
    for (int j = 0; j < 4; ++j) {
      v8bf p;
#pragma unroll
      for (int g = 0; g < 8; ++g) p[g] = (__bf16)rc[j][g];
      *(v8bf*)(xt + (size_t)(c4 + j) * 32 + kb) = p;   // 16B LDS store
    }
    __syncthreads();

    // A fragment (from L2-resident kpT) and ALL 8 B fragments up-front in
    // distinct registers: the 16 ds_load_b128 issue as one clause and drain
    // with staggered s_wait_dscnt instead of a full wait before every WMMA.
    const __bf16* abase = kpT + (size_t)(mtile * 16) * NPAD + (n0 + ks);
    v16bf a = load_a_frag(abase, NPAD, lane);
    v16bf bfr[8];
#pragma unroll
    for (int tt = 0; tt < 8; ++tt)
      bfr[tt] = load_b_frag(xt + (size_t)((cg * 8 + tt) * 16) * 32, 32, lane);
#pragma unroll
    for (int tt = 0; tt < 8; ++tt)
      acc[tt] = wmma_bf16(a, bfr[tt], acc[tt]);
    __syncthreads();
  }

  // Write f32 partials: kvp[chunk][b][m][c]
  int colb = lane & 15;
  int mh = (lane >> 4) * 8;
  float* outp = kvp + (((size_t)chunk * B_ + b) * M_) * C_;
#pragma unroll
  for (int tt = 0; tt < 8; ++tt) {
    int ct = cg * 8 + tt;
#pragma unroll
    for (int r = 0; r < 8; ++r) {
      int m = mtile * 16 + mh + r;
      outp[(size_t)m * C_ + ct * 16 + colb] = acc[tt][r];
    }
  }
}

// Deterministic reduce of partials -> kvT bf16 [b][c][m]
__global__ __launch_bounds__(256) void kv_reduce(const float* __restrict__ kvp,
                                                 __bf16* __restrict__ kvT) {
  int idx = blockIdx.x * 256 + threadIdx.x;  // < B*M*C = 262144
  int c = idx & 255;
  int m = (idx >> 8) & 63;
  int b = idx >> 14;
  float s = 0.f;
#pragma unroll 8
  for (int ch = 0; ch < NCHUNKS; ++ch)
    s += kvp[(((size_t)ch * B_ + b) * M_ + m) * C_ + c];
  kvT[((size_t)b * C_ + c) * M_ + m] = (__bf16)s;
}

// ---------------- out = (q' kv) * denom^{-1}  ----------------
// grid (157, B). Block = 8 waves; wave handles one 16-row tile x full C=256.
__global__ __launch_bounds__(256) void num_gemm(const __bf16* __restrict__ qpb,
                                                const __bf16* __restrict__ kvT,
                                                const float* __restrict__ dinv,
                                                float* __restrict__ out) {
  __shared__ __bf16 kvs[C_ * M_];  // 32 KB, [c][m]
  int b = blockIdx.y;
  int t = threadIdx.x, lane = t & 31, w = t >> 5;

  // Stage kvT[b] into LDS (16384 bf16 = 2048 uint4)
  const uint4* src = (const uint4*)(kvT + (size_t)b * C_ * M_);
  uint4* dst = (uint4*)kvs;
#pragma unroll
  for (int i = 0; i < 8; ++i) dst[t + i * 256] = src[t + i * 256];
  __syncthreads();

  int nt = blockIdx.x * 8 + w;  // 16-row tile index
  if (nt < N_ / 16) {
    const __bf16* abase = qpb + (size_t)(nt * 16) * M_;
    v16bf a0 = load_a_frag(abase, M_, lane);       // K = 0..31
    v16bf a1 = load_a_frag(abase + 32, M_, lane);  // K = 32..63

    int colb = lane & 15;
    int mh = (lane >> 4) * 8;
    float rd[8];
#pragma unroll
    for (int r = 0; r < 8; ++r) rd[r] = dinv[nt * 16 + mh + r];

    float* ob = out + ((size_t)b * N_ + (size_t)nt * 16) * C_;
#pragma unroll
    for (int ct = 0; ct < 16; ++ct) {
      v8f acc = (v8f){0.f, 0.f, 0.f, 0.f, 0.f, 0.f, 0.f, 0.f};
      const __bf16* bb = kvs + (size_t)(ct * 16) * M_;
      v16bf b0 = load_b_frag(bb, M_, lane);        // K 0..31
      v16bf b1 = load_b_frag(bb + 32, M_, lane);   // K 32..63
      acc = wmma_bf16(a0, b0, acc);
      acc = wmma_bf16(a1, b1, acc);
#pragma unroll
      for (int r = 0; r < 8; ++r)
        ob[(size_t)(mh + r) * C_ + ct * 16 + colb] = acc[r] * rd[r];
    }
  }
}

// ---------------- launch ----------------
extern "C" void kernel_launch(void* const* d_in, const int* in_sizes, int n_in,
                              void* d_out, int out_size, void* d_ws, size_t ws_size,
                              hipStream_t stream) {
  const float* x    = (const float*)d_in[0];  // [B,N,C]
  const float* nv1  = (const float*)d_in[1];  // [N,D]
  const float* nv2  = (const float*)d_in[2];  // [N,D]
  const float* proj = (const float*)d_in[3];  // [M,D]

  float* out_main = (float*)d_out;                                   // [B,N,C]
  float* out_q    = out_main + (size_t)B_ * N_ * C_;                 // [N,M]
  float* out_k    = out_q + (size_t)N_ * M_;                         // [N,M]

  // Carve workspace (256B-aligned slabs)
  char* p = (char*)d_ws;
  auto carve = [&](size_t bytes) {
    void* r = (void*)p;
    p += (bytes + 255) & ~(size_t)255;
    return r;
  };
  __bf16* kpT   = (__bf16*)carve((size_t)M_ * NPAD * 2);                   // [M][NPAD]
  __bf16* qpb   = (__bf16*)carve((size_t)N_ * M_ * 2);                     // [N][M]
  float*  kdash = (float*)carve((size_t)N_ * M_ * 4);
  float*  diagk = (float*)carve((size_t)N_ * 4);
  float*  bmax  = (float*)carve((size_t)(N_ / 4) * 4);
  float*  kstab = (float*)carve(256);
  float*  kpart = (float*)carve((size_t)(NPAD * 64 / 256) * 64 * 4);
  float*  ksum  = (float*)carve(64 * 4);
  float*  dinv  = (float*)carve((size_t)N_ * 4);
  float*  kvp   = (float*)carve((size_t)NCHUNKS * B_ * M_ * C_ * 4);
  __bf16* kvT   = (__bf16*)carve((size_t)B_ * C_ * M_ * 2);

  // Feature maps
  fk1_kdash<<<N_ / 4, 256, 0, stream>>>(nv2, proj, kdash, diagk, bmax);
  fk2_max<<<1, 256, 0, stream>>>(bmax, kstab, N_ / 4);
  fk3_kprime<<<NPAD * 64 / 256, 256, 0, stream>>>(kdash, diagk, kstab, out_k, kpT, kpart);
  fk4_ksum<<<1, 256, 0, stream>>>(kpart, ksum);
  fq_qprime<<<N_ / 4, 256, 0, stream>>>(nv1, proj, ksum, out_q, qpb, dinv);

  // kv = k'^T x   (split-K, then reduce)
  kv_gemm<<<dim3(NCHUNKS, B_), 256, 0, stream>>>(x, kpT, kvp);
  kv_reduce<<<(B_ * M_ * C_) / 256, 256, 0, stream>>>(kvp, kvT);

  // out = (q' kv) / denom
  num_gemm<<<dim3((N_ / 16 + 7) / 8, B_), 256, 0, stream>>>(qpb, kvT, dinv, out_main);
}
